// FlashAttention_6055903887815
// MI455X (gfx1250) — matrix operand order repaired and checked by reading, not compiled, not run on hardware
//
#include <hip/hip_runtime.h>
#include <stdint.h>

// ---------- CDNA5 WMMA / TDM types ----------
typedef __attribute__((ext_vector_type(16))) __bf16 bf16x16;
typedef __attribute__((ext_vector_type(8)))  float  f32x8;
typedef __attribute__((ext_vector_type(4)))  float  f32x4;
typedef __attribute__((ext_vector_type(4)))  unsigned int u32x4;
typedef __attribute__((ext_vector_type(8)))  int    i32x8;
typedef __attribute__((ext_vector_type(4)))  int    i32x4;

union ABFrag { bf16x16 v; u32x4 u[2]; };
union Pack8  { uint16_t h[8]; u32x4 u; };

__device__ __forceinline__ uint16_t f2bf(float f) {
  union { float f; unsigned int u; } x; x.f = f;
  unsigned int r = x.u + 0x7FFFu + ((x.u >> 16) & 1u);   // round-to-nearest-even
  return (uint16_t)(r >> 16);
}

__device__ __forceinline__ f32x8 wmma_bf16(const ABFrag& a, const ABFrag& b, f32x8 c) {
  return __builtin_amdgcn_wmma_f32_16x16x32_bf16(false, a.v, false, b.v, (short)0, c,
                                                 false, false);
}

__device__ __forceinline__ float rowmax16(float v) {
#pragma unroll
  for (int m = 8; m >= 1; m >>= 1) v = fmaxf(v, __shfl_xor(v, m, 32));
  return v;
}
__device__ __forceinline__ float rowsum16(float v) {
#pragma unroll
  for (int m = 8; m >= 1; m >>= 1) v += __shfl_xor(v, m, 32);
  return v;
}

// -------- Tensor Data Mover: 2-D bf16 tile (tile_w x tile_h) -> LDS ---------
// D# layout per CDNA5 ISA ch.8: group0 = {count, lds_addr, global_addr, type=2},
// group1 = {data_size=1(2B), tensor_dim0/1, tile_dim0/1, dim0_stride}.
__device__ __forceinline__ void tdm_load_2d_bf16(
    unsigned lds_off, const void* gaddr, unsigned tile_w, unsigned tile_h,
    unsigned row_stride_elems, unsigned tensor_w, unsigned tensor_h) {
  unsigned long long ga = (unsigned long long)gaddr;
  u32x4 g0;
  g0.x = 1u;                                              // count=1 (valid D#)
  g0.y = lds_off;                                         // lds_addr (bytes)
  g0.z = (unsigned)(ga & 0xFFFFFFFFu);                    // global_addr[31:0]
  g0.w = (unsigned)((ga >> 32) & 0x01FFFFFFu) | (2u << 30); // addr[56:32]|type=2
  i32x8 g1;
  g1[0] = (int)(1u << 16);                                // wg_mask=0, data_size=2B
  g1[1] = (int)((tensor_w & 0xFFFFu) << 16);              // tensor_dim0[15:0]
  g1[2] = (int)(((tensor_w >> 16) & 0xFFFFu) | ((tensor_h & 0xFFFFu) << 16));
  g1[3] = (int)(((tensor_h >> 16) & 0xFFFFu) | ((tile_w & 0xFFFFu) << 16));
  g1[4] = (int)(tile_h & 0xFFFFu);                        // tile_dim1; tile_dim2=0
  g1[5] = (int)row_stride_elems;                          // dim0_stride[31:0]
  g1[6] = 0;
  g1[7] = 0;
  i32x4 z4 = {0, 0, 0, 0};
#if __clang_major__ >= 23
  i32x8 z8 = {0, 0, 0, 0, 0, 0, 0, 0};
  __builtin_amdgcn_tensor_load_to_lds(g0, g1, z4, z4, z8, 0);
#else
  __builtin_amdgcn_tensor_load_to_lds(g0, g1, z4, z4, 0);
#endif
}

// Problem constants
#define BATCH 4
#define SEQ   2048
#define DIMC  1024
#define HEADS 16
#define HDIM  64
#define NOUT  3072
#define MROWS 8192
#define QSCALE 0.015625f   // (1/8)*(1/8): q*scale then SDPA's own 1/sqrt(D)

// ---------------------------------------------------------------------------
// Kernel 0: fp32 weights -> bf16, transposed k-major for B-frag loads.
// ---------------------------------------------------------------------------
__global__ __launch_bounds__(256) void convert_weights_kernel(
    const float* __restrict__ qkv_w, const float* __restrict__ proj_w,
    uint16_t* __restrict__ Wqt, uint16_t* __restrict__ Wpt) {
  const int idx = blockIdx.x * 256 + threadIdx.x;
  if (idx < NOUT * DIMC) {
    const int o = idx >> 10, k = idx & (DIMC - 1);
    Wqt[(size_t)k * NOUT + o] = f2bf(qkv_w[idx]);
  } else {
    const int j = idx - NOUT * DIMC;
    const int o = j >> 10, k = j & (DIMC - 1);
    Wpt[(size_t)k * DIMC + o] = f2bf(proj_w[j]);
  }
}

// ---------------------------------------------------------------------------
// Kernel 1: QKV GEMM (64x128 block tile, 8 waves). W tiles streamed by the
// Tensor Data Mover, double buffered; x tile staged cooperatively with fp32
// -> bf16 conversion. Epilogue routes (scalarized) to Q(scaled)/K^T/V.
// ---------------------------------------------------------------------------
__global__ __launch_bounds__(256) void qkv_gemm_kernel(
    const float* __restrict__ x, const uint16_t* __restrict__ Wqt,
    const float* __restrict__ q_bias, const float* __restrict__ v_bias,
    uint16_t* __restrict__ Qb, uint16_t* __restrict__ Ktb,
    uint16_t* __restrict__ Vb) {
  __shared__ uint16_t lx[64][32];        // A tile, k contiguous
  __shared__ uint16_t lw[2][32][128];    // B tile double buffer (TDM dest)

  const int tid  = threadIdx.x;
  const int lane = tid & 31;
  const int w    = tid >> 5;
  const int m0   = blockIdx.y * 64;
  const int n0   = blockIdx.x * 128;
  const int wm   = (w & 3) * 16;
  const int wn   = (w >> 2) * 64;
  const int hl   = lane >> 4;
  const int ln   = lane & 15;
  const int c0   = hl * 8;

  const int xr = tid >> 2;
  const int xk = (tid & 3) * 8;

  f32x8 acc[4] = {};

  // prologue: TDM fetch of W tile 0
  if (w == 0) {
    tdm_load_2d_bf16((unsigned)(size_t)&lw[0][0][0], Wqt + n0,
                     128, 32, NOUT, NOUT, DIMC);
  }

  for (int t = 0; t < DIMC / 32; ++t) {
    const int k0  = t * 32;
    const int buf = t & 1;
    // stage x tile (fp32 -> bf16)
    const float* xp = x + (size_t)(m0 + xr) * DIMC + k0 + xk;
    f32x4 a0 = *(const f32x4*)xp;
    f32x4 a1 = *(const f32x4*)(xp + 4);
    Pack8 pk;
#pragma unroll
    for (int i = 0; i < 4; i++) { pk.h[i] = f2bf(a0[i]); pk.h[4 + i] = f2bf(a1[i]); }
    *(u32x4*)&lx[xr][xk] = pk.u;

    if (w == 0) {
      if (t + 1 < DIMC / 32) {
        tdm_load_2d_bf16((unsigned)(size_t)&lw[buf ^ 1][0][0],
                         Wqt + (size_t)(k0 + 32) * NOUT + n0,
                         128, 32, NOUT, NOUT, DIMC);
        __builtin_amdgcn_s_wait_tensorcnt(1);   // all but newest complete
      } else {
        __builtin_amdgcn_s_wait_tensorcnt(0);
      }
    }
    __syncthreads();

    ABFrag af;
    af.u[0] = *(const u32x4*)&lx[wm + ln][c0];
    af.u[1] = *(const u32x4*)&lx[wm + ln][c0 + 16];
    ABFrag bfr[4];
#pragma unroll
    for (int f = 0; f < 4; f++) {
      for (int s = 0; s < 16; s++)
        bfr[f].v[s] = __builtin_bit_cast(__bf16, lw[buf][c0 + (s & 7) + ((s >> 3) << 4)][wn + f * 16 + ln]);
    }
#pragma unroll
    for (int f = 0; f < 4; f++) acc[f] = wmma_bf16(af, bfr[f], acc[f]);
    __syncthreads();
  }

  // epilogue: scalarized routing (p,h are wave-uniform per fragment)
#pragma unroll
  for (int f = 0; f < 4; f++) {
    const int o  = n0 + wn + f * 16 + ln;
    const int ou = __builtin_amdgcn_readfirstlane(o);   // lane0 => ln==0
    const int p  = ou >> 10;
    const int h  = (ou & (DIMC - 1)) >> 6;
    const int c  = o & (DIMC - 1);
    const int d  = c & 63;
    if (p == 0) {
      const float qb = q_bias[c];
#pragma unroll
      for (int r = 0; r < 8; r++) {
        const int mg = m0 + wm + r + 8 * hl;
        const int b = mg >> 11, nr = mg & (SEQ - 1);
        Qb[(((size_t)(b * HEADS + h) * SEQ + nr) << 6) + d] =
            f2bf((acc[f][r] + qb) * QSCALE);
      }
    } else if (p == 1) {
#pragma unroll
      for (int r = 0; r < 8; r++) {
        const int mg = m0 + wm + r + 8 * hl;
        const int b = mg >> 11, nr = mg & (SEQ - 1);
        Ktb[(((size_t)(b * HEADS + h) * HDIM + d) << 11) + nr] = f2bf(acc[f][r]);
      }
    } else {
      const float vb = v_bias[c];
#pragma unroll
      for (int r = 0; r < 8; r++) {
        const int mg = m0 + wm + r + 8 * hl;
        const int b = mg >> 11, nr = mg & (SEQ - 1);
        Vb[(((size_t)(b * HEADS + h) * SEQ + nr) << 6) + d] =
            f2bf(acc[f][r] + vb);
      }
    }
  }
}

// ---------------------------------------------------------------------------
// Kernel 2: flash attention. One wave per 16-query tile per (b,h).
// V loads are issued before the softmax VALU block to overlap; per-wave
// private LDS P-transpose needs no barriers (DS ops in-order per wave).
// ---------------------------------------------------------------------------
__global__ __launch_bounds__(256) void attn_kernel(
    const uint16_t* __restrict__ Qb, const uint16_t* __restrict__ Ktb,
    const uint16_t* __restrict__ Vb, uint16_t* __restrict__ Ao) {
  __shared__ uint16_t lp[8][16][32];

  const int tid  = threadIdx.x;
  const int lane = tid & 31;
  const int w    = tid >> 5;
  const int gw   = blockIdx.x * 8 + w;
  const int qt   = gw & 127;
  const int bh   = gw >> 7;
  const int q0   = qt * 16;
  const int hl   = lane >> 4;
  const int ln   = lane & 15;
  const int c0   = hl * 8;

  const uint16_t* Qbase = Qb  + (size_t)bh * SEQ * HDIM;
  const uint16_t* Kbase = Ktb + (size_t)bh * HDIM * SEQ;
  const uint16_t* Vbase = Vb  + (size_t)bh * SEQ * HDIM;

  ABFrag qa0, qa1;
  {
    const uint16_t* qp = Qbase + (size_t)(q0 + ln) * HDIM;
    qa0.u[0] = *(const u32x4*)(qp + c0);
    qa0.u[1] = *(const u32x4*)(qp + c0 + 16);
    qa1.u[0] = *(const u32x4*)(qp + 32 + c0);
    qa1.u[1] = *(const u32x4*)(qp + 32 + c0 + 16);
  }

  f32x8 o0 = {}, o1 = {}, o2 = {}, o3 = {};
  float mrow[8], lrow[8];
#pragma unroll
  for (int r = 0; r < 8; r++) { mrow[r] = -1e30f; lrow[r] = 0.f; }

  for (int kb = 0; kb < SEQ; kb += 32) {
    // ---- K B-frags + QK^T WMMAs ----
    f32x8 s0 = {}, s1 = {};
    ABFrag bk[4];
#pragma unroll
    for (int dc = 0; dc < 2; dc++) {
      for (int j = 0; j < 2; j++)
        for (int s = 0; s < 16; s++)
          bk[2 * dc + j].v[s] = __builtin_bit_cast(__bf16,
              Kbase[(size_t)(dc * 32 + c0 + (s & 7) + ((s >> 3) << 4)) * SEQ + kb + j * 16 + ln]);
    }
    // ---- issue V loads early: they complete under the softmax VALU work ----
    const uint16_t* vp = Vbase + (size_t)(kb + lane) * HDIM;
    ABFrag bv[4];
#pragma unroll
    for (int f = 0; f < 4; f++) {
      for (int s = 0; s < 16; s++)
        bv[f].v[s] = __builtin_bit_cast(__bf16,
            vp[(size_t)(c0 + (s & 7) + ((s >> 3) << 4) - lane) * HDIM + f * 16 + ln]);
    }
    if (kb + 32 < SEQ) {     // prefetch next K/V chunk (-> global_prefetch_b8)
      __builtin_prefetch(Kbase + (size_t)lane * SEQ + kb + 32, 0, 3);
      __builtin_prefetch(Vbase + (size_t)(kb + 32 + lane) * HDIM, 0, 3);
    }
    s0 = wmma_bf16(qa0, bk[0], s0);
    s1 = wmma_bf16(qa0, bk[1], s1);
    s0 = wmma_bf16(qa1, bk[2], s0);
    s1 = wmma_bf16(qa1, bk[3], s1);

    // ---- online softmax; P -> LDS (bf16) ----
#pragma unroll
    for (int r = 0; r < 8; r++) {
      float rv = rowmax16(fmaxf(s0[r], s1[r]));
      float mn = fmaxf(mrow[r], rv);
      float al = __expf(mrow[r] - mn);
      float p0 = __expf(s0[r] - mn);
      float p1 = __expf(s1[r] - mn);
      float rs = rowsum16(p0 + p1);
      lrow[r] = lrow[r] * al + rs;
      mrow[r] = mn;
      o0[r] *= al; o1[r] *= al; o2[r] *= al; o3[r] *= al;
      lp[w][r + 8 * hl][ln]      = f2bf(p0);
      lp[w][r + 8 * hl][ln + 16] = f2bf(p1);
    }

    // ---- P back as A-frag (same-wave LDS: in-order, no barrier) ----
    ABFrag pf;
    pf.u[0] = *(const u32x4*)&lp[w][ln][c0];
    pf.u[1] = *(const u32x4*)&lp[w][ln][c0 + 16];

    // ---- O += P @ V ----
    o0 = wmma_bf16(pf, bv[0], o0);
    o1 = wmma_bf16(pf, bv[1], o1);
    o2 = wmma_bf16(pf, bv[2], o2);
    o3 = wmma_bf16(pf, bv[3], o3);
  }

  const int b = bh >> 4, h = bh & 15;
#pragma unroll
  for (int r = 0; r < 8; r++) {
    const float inv = 1.0f / lrow[r];
    uint16_t* op = Ao + (size_t)(b * SEQ + q0 + r + 8 * hl) * DIMC + h * HDIM + ln;
    op[0]  = f2bf(o0[r] * inv);
    op[16] = f2bf(o1[r] * inv);
    op[32] = f2bf(o2[r] * inv);
    op[48] = f2bf(o3[r] * inv);
  }
}

// ---------------------------------------------------------------------------
// Kernel 3: output projection (fp32 out + bias). Same tiling; W via TDM.
// ---------------------------------------------------------------------------
__global__ __launch_bounds__(256) void proj_gemm_kernel(
    const uint16_t* __restrict__ Ao, const uint16_t* __restrict__ Wpt,
    const float* __restrict__ proj_b, float* __restrict__ out) {
  __shared__ uint16_t la[64][32];
  __shared__ uint16_t lw[2][32][128];

  const int tid  = threadIdx.x;
  const int lane = tid & 31;
  const int w    = tid >> 5;
  const int m0   = blockIdx.y * 64;
  const int n0   = blockIdx.x * 128;
  const int wm   = (w & 3) * 16;
  const int wn   = (w >> 2) * 64;
  const int hl   = lane >> 4;
  const int ln   = lane & 15;
  const int c0   = hl * 8;

  const int xr = tid >> 2;
  const int xk = (tid & 3) * 8;

  f32x8 acc[4] = {};

  if (w == 0) {
    tdm_load_2d_bf16((unsigned)(size_t)&lw[0][0][0], Wpt + n0,
                     128, 32, DIMC, DIMC, DIMC);
  }

  for (int t = 0; t < DIMC / 32; ++t) {
    const int k0  = t * 32;
    const int buf = t & 1;
    *(u32x4*)&la[xr][xk] =
        *(const u32x4*)(Ao + (size_t)(m0 + xr) * DIMC + k0 + xk);
    if (w == 0) {
      if (t + 1 < DIMC / 32) {
        tdm_load_2d_bf16((unsigned)(size_t)&lw[buf ^ 1][0][0],
                         Wpt + (size_t)(k0 + 32) * DIMC + n0,
                         128, 32, DIMC, DIMC, DIMC);
        __builtin_amdgcn_s_wait_tensorcnt(1);
      } else {
        __builtin_amdgcn_s_wait_tensorcnt(0);
      }
    }
    __syncthreads();

    ABFrag af;
    af.u[0] = *(const u32x4*)&la[wm + ln][c0];
    af.u[1] = *(const u32x4*)&la[wm + ln][c0 + 16];
    ABFrag bfr[4];
#pragma unroll
    for (int f = 0; f < 4; f++) {
      for (int s = 0; s < 16; s++)
        bfr[f].v[s] = __builtin_bit_cast(__bf16, lw[buf][c0 + (s & 7) + ((s >> 3) << 4)][wn + f * 16 + ln]);
    }
#pragma unroll
    for (int f = 0; f < 4; f++) acc[f] = wmma_bf16(af, bfr[f], acc[f]);
    __syncthreads();
  }

#pragma unroll
  for (int f = 0; f < 4; f++) {
    const int o = n0 + wn + f * 16 + ln;
    const float bias = proj_b[o];
#pragma unroll
    for (int r = 0; r < 8; r++) {
      const int mg = m0 + wm + r + 8 * hl;
      out[(size_t)mg * DIMC + o] = acc[f][r] + bias;
    }
  }
}

// ---------------------------------------------------------------------------
extern "C" void kernel_launch(void* const* d_in, const int* in_sizes, int n_in,
                              void* d_out, int out_size, void* d_ws, size_t ws_size,
                              hipStream_t stream) {
  (void)in_sizes; (void)n_in; (void)out_size; (void)ws_size;
  const float* x      = (const float*)d_in[0];
  const float* qkv_w  = (const float*)d_in[1];
  const float* q_bias = (const float*)d_in[2];
  const float* v_bias = (const float*)d_in[3];
  const float* proj_w = (const float*)d_in[4];
  const float* proj_b = (const float*)d_in[5];
  float* out = (float*)d_out;

  char* ws = (char*)d_ws;                                  // 72 MB total
  uint16_t* Qb  = (uint16_t*)(ws);                         // 16 MB [B,H,N,D]
  uint16_t* Ktb = (uint16_t*)(ws + (16u << 20));           // 16 MB [B,H,D,N]
  uint16_t* Vb  = (uint16_t*)(ws + (32u << 20));           // 16 MB [B,H,N,D]
  uint16_t* Wqt = (uint16_t*)(ws + (48u << 20));           //  6 MB [C,3C]
  uint16_t* Wpt = (uint16_t*)(ws + (54u << 20));           //  2 MB [C,C]
  uint16_t* Ao  = (uint16_t*)(ws + (56u << 20));           // 16 MB [B,N,C]

  convert_weights_kernel<<<(NOUT * DIMC + DIMC * DIMC) / 256, 256, 0, stream>>>(
      qkv_w, proj_w, Wqt, Wpt);
  qkv_gemm_kernel<<<dim3(NOUT / 128, MROWS / 64), 256, 0, stream>>>(
      x, Wqt, q_bias, v_bias, Qb, Ktb, Vb);
  attn_kernel<<<(BATCH * HEADS * (SEQ / 16)) / 8, 256, 0, stream>>>(
      Qb, Ktb, Vb, Ao);
  proj_gemm_kernel<<<dim3(DIMC / 128, MROWS / 64), 256, 0, stream>>>(
      Ao, Wpt, proj_b, out);
}
